// DotProductAttention_91130616087216
// MI455X (gfx1250) — matrix operand order repaired and checked by reading, not compiled, not run on hardware
//
#include <hip/hip_runtime.h>

// Flash-attention, B=16, TQ=TK=2048, D=DV=128, fp32 I/O, per-batch key-length
// masking. Math: v_wmma_f32_16x16x32_bf16 with f32 accumulation + online
// softmax. Compute-bound (~34 GFLOP vs ~3us of HBM traffic at 23.3 TB/s).
//
// Fast path (needs 16 MiB of d_ws): K and V are pre-converted to bf16 once
// (V pre-transposed per 32-key tile => contiguous 8KB tile chunks). The hot
// loop double-buffers tiles into LDS with the Tensor Data Mover
// (tensor_load_to_lds, TENSORcnt) -- zero VGPR staging, one barrier/iter.
// Row-sums of P come from an extra WMMA against an all-ones B fragment;
// 1/sqrt(D) is folded into the Q->bf16 staging; only the tail tile masks.

#define TQ_ 2048
#define TK_ 2048
#define D_  128
#define DV_ 128
#define B_  16
#define MASK_VAL -1000000.0f

typedef __attribute__((ext_vector_type(16))) __bf16 bf16x16;
typedef __attribute__((ext_vector_type(8)))  float  f32x8;
typedef __attribute__((ext_vector_type(4)))  unsigned u32x4;
typedef __attribute__((ext_vector_type(8)))  int      i32x8;
typedef __attribute__((ext_vector_type(4)))  int      i32x4;

#if defined(__HIP_DEVICE_COMPILE__) && __has_builtin(__builtin_amdgcn_tensor_load_to_lds)
#define HAVE_TDM 1
#else
#define HAVE_TDM 0
#endif

union Frag {
    bf16x16  v;
    unsigned u[8];
    uint4    q[2];
};

__device__ __forceinline__ unsigned short f2bf(float f) {
    unsigned u = __float_as_uint(f);
    u += 0x7FFFu + ((u >> 16) & 1u);   // round-to-nearest-even
    return (unsigned short)(u >> 16);
}
__device__ __forceinline__ unsigned packbf(float lo, float hi) {
    return (unsigned)f2bf(lo) | ((unsigned)f2bf(hi) << 16);
}

#if HAVE_TDM
// DMA one contiguous 4KB chunk (512 x 8-byte units) from global to LDS.
// D# per CDNA5 ISA: group0 = count/lds_addr/global_addr/type,
// group1 = data_size + dims, groups 2&3 unused (<=2D tensor).
__device__ __forceinline__ void tdm_load_4kb(unsigned lds_off, const void* gsrc) {
    unsigned long long ga = (unsigned long long)gsrc;
    u32x4 g0 = { 1u,                                   // count=1, user mode
                 lds_off,                              // lds_addr (bytes)
                 (unsigned)ga,                         // global_addr[31:0]
                 ((unsigned)(ga >> 32) & 0x01FFFFFFu)  // global_addr[56:32]
                   | 0x80000000u };                    // type=2 (bits 127:126)
    i32x8 g1 = { 0x30000,            // workgroup_mask=0, data_size=3 (8B)
                 (int)(512u << 16),  // tensor_dim0 = 512 (bits 63:48)
                 (int)(1u   << 16),  // tensor_dim1 = 1   (bits 95:80)
                 (int)(512u << 16),  // tile_dim0   = 512 (bits 127:112)
                 1,                  // tile_dim1 = 1, tile_dim2 = 0
                 512,                // tensor_dim0_stride = 512
                 0, 0 };
    i32x4 z4 = { 0, 0, 0, 0 };
#if __clang_major__ >= 23
    i32x8 z8 = { 0, 0, 0, 0, 0, 0, 0, 0 };
    __builtin_amdgcn_tensor_load_to_lds(g0, g1, z4, z4, z8, 0);
#else
    __builtin_amdgcn_tensor_load_to_lds(g0, g1, z4, z4, 0);
#endif
}
#endif

// ---------------------------------------------------------------------------
// One 32-key tile: S = Q K^T (8 WMMAs), online softmax, O += P V (8 WMMAs),
// row-sums via one WMMA against all-ones B. MASKED only for the tail tile.
// ---------------------------------------------------------------------------
template<bool MASKED>
__device__ __forceinline__ void tile_body(
        int t, const unsigned short* __restrict__ Kl,
        const unsigned short* __restrict__ Vtl, unsigned short* __restrict__ lP,
        const Frag* __restrict__ qa, const Frag& ones,
        f32x8* __restrict__ o, float* __restrict__ mrow, float* __restrict__ lrow,
        int wave, int lane, int vlen)
{
    const float LOG2E = 1.4426950408889634f;
    const int n = lane & 15;

    // ---- S = Q K^T: two 16x16 subtiles, 4 WMMAs each ----------------------
    f32x8 s0, s1;
    #pragma unroll
    for (int r = 0; r < 8; ++r) { s0[r] = 0.0f; s1[r] = 0.0f; }
    {
        int ko = (lane < 16) ? 0 : 8;
        #pragma unroll
        for (int c = 0; c < 4; ++c) {
            Frag kb0, kb1;
            const uint4* p0 = (const uint4*)&Kl[(n)      * D_ + c * 32 + ko];
            const uint4* p1 = (const uint4*)&Kl[(16 + n) * D_ + c * 32 + ko];
            kb0.q[0] = p0[0]; kb0.q[1] = p0[2];
            kb1.q[0] = p1[0]; kb1.q[1] = p1[2];
            s0 = __builtin_amdgcn_wmma_f32_16x16x32_bf16(
                    false, qa[c].v, false, kb0.v, (short)0, s0, false, false);
            s1 = __builtin_amdgcn_wmma_f32_16x16x32_bf16(
                    false, qa[c].v, false, kb1.v, (short)0, s1, false, false);
        }
    }

    // ---- mask (tail tile only; Q was pre-scaled by 1/sqrt(D)) -------------
    if (MASKED) {
        const bool ok0 = (t * 32 + n)      < vlen;
        const bool ok1 = (t * 32 + 16 + n) < vlen;
        #pragma unroll
        for (int r = 0; r < 8; ++r) {
            s0[r] = ok0 ? s0[r] : MASK_VAL;
            s1[r] = ok1 ? s1[r] : MASK_VAL;
        }
    }

    // ---- row max, alpha, P = exp(S - m) ----------------------------------
    float al[8];
    #pragma unroll
    for (int r = 0; r < 8; ++r) {
        float mx = fmaxf(s0[r], s1[r]);
        #pragma unroll
        for (int off = 8; off >= 1; off >>= 1)
            mx = fmaxf(mx, __shfl_xor(mx, off, 32));   // within 16-lane group
        float mnew = fmaxf(mrow[r], mx);
        al[r] = __builtin_amdgcn_exp2f((mrow[r] - mnew) * LOG2E);
        float p0 = __builtin_amdgcn_exp2f((s0[r] - mnew) * LOG2E);
        float p1 = __builtin_amdgcn_exp2f((s1[r] - mnew) * LOG2E);
        int row = r + ((lane >> 4) << 3);
        lP[(wave * 16 + row) * 32 + n]      = f2bf(p0);
        lP[(wave * 16 + row) * 32 + 16 + n] = f2bf(p1);
        mrow[r] = mnew;
    }

    // ---- reload P as A-fragment ------------------------------------------
    Frag pa;
    {
        int m  = lane & 15;
        int kb = (lane < 16) ? 0 : 8;
        int base = (wave * 16 + m) * 32;
        #pragma unroll
        for (int j = 0; j < 8; ++j) {
            int k = (j < 4) ? (kb + 2 * j) : (16 + kb + 2 * (j - 4));
            pa.u[j] = *(const unsigned*)&lP[base + k];
        }
    }

    // ---- row sums in one WMMA: rs = P @ ones(32x16) ----------------------
    {
        f32x8 zero;
        #pragma unroll
        for (int r = 0; r < 8; ++r) zero[r] = 0.0f;
        f32x8 rs = __builtin_amdgcn_wmma_f32_16x16x32_bf16(
                      false, pa.v, false, ones.v, (short)0, zero, false, false);
        #pragma unroll
        for (int r = 0; r < 8; ++r) lrow[r] = lrow[r] * al[r] + rs[r];
    }

    // ---- O = diag(alpha)*O + P @ V : 8 WMMAs across DV --------------------
    {
        int ko = (lane < 16) ? 0 : 8;
        #pragma unroll
        for (int c = 0; c < 8; ++c) {
            f32x8 oc = o[c];
            #pragma unroll
            for (int r = 0; r < 8; ++r) oc[r] *= al[r];
            Frag vb;
            const uint4* pv = (const uint4*)&Vtl[(c * 16 + n) * 32 + ko];
            vb.q[0] = pv[0]; vb.q[1] = pv[2];
            o[c] = __builtin_amdgcn_wmma_f32_16x16x32_bf16(
                    false, pa.v, false, vb.v, (short)0, oc, false, false);
        }
    }
}

// ---------------------------------------------------------------------------
// Prepass 1: K fp32 -> bf16, row-major (tile t = contiguous 8KB chunk).
// ---------------------------------------------------------------------------
__global__ __launch_bounds__(256)
void conv_k_kernel(const float* __restrict__ src, unsigned short* __restrict__ dst, int n4) {
    int i = blockIdx.x * blockDim.x + threadIdx.x;
    if (i < n4) {
        float4 f = ((const float4*)src)[i];
        uint2 p;
        p.x = packbf(f.x, f.y);
        p.y = packbf(f.z, f.w);
        ((uint2*)dst)[i] = p;
    }
}

// ---------------------------------------------------------------------------
// Prepass 2: V fp32 [b][key][dv] -> bf16 [b][tile][dv][key%32] (contiguous 8KB
// per 32-key tile, already in WMMA-B-friendly transposed layout).
// ---------------------------------------------------------------------------
__global__ __launch_bounds__(256)
void conv_vt_kernel(const float* __restrict__ V, unsigned short* __restrict__ Vt) {
    const int b = blockIdx.y, t = blockIdx.x;
    const float* src = V + ((size_t)b * TK_ + (size_t)t * 32) * DV_;
    unsigned short* dst = Vt + ((size_t)b * (TK_ / 32) + t) * (DV_ * 32);
    #pragma unroll
    for (int i = 0; i < 4; ++i) {
        int idx4 = i * 256 + threadIdx.x;        // float4 index over 32x128
        float4 f = ((const float4*)src)[idx4];
        int key = idx4 >> 5;                     // 32 float4 per key row
        int dv  = (idx4 & 31) * 4;
        dst[(dv + 0) * 32 + key] = f2bf(f.x);
        dst[(dv + 1) * 32 + key] = f2bf(f.y);
        dst[(dv + 2) * 32 + key] = f2bf(f.z);
        dst[(dv + 3) * 32 + key] = f2bf(f.w);
    }
}

// ---------------------------------------------------------------------------
// Fast attention kernel: bf16 K/Vt from workspace, TDM double-buffered tiles.
// LDS: Q 16KB + 2x(K 8KB + Vt 8KB) + P 4KB = 52KB.
// ---------------------------------------------------------------------------
__global__ __launch_bounds__(128, 1)
void fattn_fast(const float* __restrict__ Q, const unsigned short* __restrict__ Kbf,
                const unsigned short* __restrict__ Vtbf, const int* __restrict__ vlens,
                float* __restrict__ Out)
{
    __shared__ __align__(16) unsigned short lQ[64 * D_];
    __shared__ __align__(16) unsigned short lK[2][32 * D_];
    __shared__ __align__(16) unsigned short lVt[2][DV_ * 32];
    __shared__ __align__(16) unsigned short lP[4 * 16 * 32];

    const int b     = blockIdx.y;
    const int qtile = blockIdx.x;
    const int tid   = threadIdx.x;
    const int wave  = tid >> 5;
    const int lane  = tid & 31;

    const float SCALE = 0.08838834764831845f;   // 1/sqrt(128)

    const float* Qb = Q + ((size_t)b * TQ_ + (size_t)qtile * 64) * D_;
    const unsigned short* Kb  = Kbf  + (size_t)b * TK_ * D_;
    const unsigned short* Vtb = Vtbf + (size_t)b * (TK_ / 32) * (DV_ * 32);

    // ---- stage Q tile (fp32 -> bf16, pre-scaled by 1/sqrt(D)) -------------
    for (int i = 0; i < 16; ++i) {
        int idx4 = i * 128 + tid;
        float4 f = ((const float4*)Qb)[idx4];
        int h = idx4 * 4;
        *(unsigned*)&lQ[h]     = packbf(f.x * SCALE, f.y * SCALE);
        *(unsigned*)&lQ[h + 2] = packbf(f.z * SCALE, f.w * SCALE);
    }
    __syncthreads();

    // ---- Q A-fragments (ISA 16-bit A 16x32 layout), 4 chunks of K=32 ------
    Frag qa[4];
    {
        int m  = lane & 15;
        int kb = (lane < 16) ? 0 : 8;
        for (int c = 0; c < 4; ++c) {
            int base = (wave * 16 + m) * D_ + c * 32;
            #pragma unroll
            for (int j = 0; j < 8; ++j) {
                int k = (j < 4) ? (kb + 2 * j) : (16 + kb + 2 * (j - 4));
                qa[c].u[j] = *(const unsigned*)&lQ[base + k];
            }
        }
    }

    Frag ones;   // all-ones bf16 B fragment for row sums (1.0bf16 = 0x3F80)
    #pragma unroll
    for (int j = 0; j < 8; ++j) ones.u[j] = 0x3F803F80u;

    f32x8 o[8];
    float mrow[8], lrow[8];
    #pragma unroll
    for (int c = 0; c < 8; ++c)
        #pragma unroll
        for (int r = 0; r < 8; ++r) o[c][r] = 0.0f;
    #pragma unroll
    for (int r = 0; r < 8; ++r) { mrow[r] = -3.0e38f; lrow[r] = 0.0f; }

    int vlen = vlens[b];
    if (vlen < 1) vlen = 1;
    if (vlen > TK_) vlen = TK_;
    const int nfull  = vlen >> 5;                  // fully-valid tiles
    const int rem    = vlen & 31;                  // partial tail tile?
    const int ntiles = nfull + (rem ? 1 : 0);

#if HAVE_TDM
    // Per-wave DMA split: wave 0/1 -> K tile halves, wave 2/3 -> Vt halves.
    // Scalarize wave id so descriptor math stays in SGPRs (TDM ignores EXEC).
    const int      w    = __builtin_amdgcn_readfirstlane(wave);
    const int      sel  = w >> 1;          // 0 = K, 1 = Vt
    const int      half = w & 1;
    const unsigned lK_off  = (unsigned)(unsigned long long)(void*)&lK[0][0];
    const unsigned lVt_off = (unsigned)(unsigned long long)(void*)&lVt[0][0];

    // prologue: tile 0 into buffer 0
    {
        const unsigned short* gsrc = (sel ? Vtb : Kb) + half * 2048;   // halves
        unsigned dst = (sel ? lVt_off : lK_off) + (unsigned)half * 4096;
        tdm_load_4kb(dst, gsrc);
        __builtin_amdgcn_s_wait_tensorcnt(0);
    }
    __syncthreads();
#else
    {
        const uint4* kp = (const uint4*)Kb;
        const uint4* vp = (const uint4*)Vtb;
        uint4* kd = (uint4*)&lK[0][0];
        uint4* vd = (uint4*)&lVt[0][0];
        #pragma unroll
        for (int i = 0; i < 4; ++i) { kd[i * 128 + tid] = kp[i * 128 + tid];
                                      vd[i * 128 + tid] = vp[i * 128 + tid]; }
    }
    __syncthreads();
#endif

    for (int t = 0; t < nfull; ++t) {
        const bool have_next = (t + 1 < ntiles);
#if HAVE_TDM
        if (have_next) {
            const int bb = (t + 1) & 1;
            const unsigned short* gsrc =
                (sel ? (Vtb + (size_t)(t + 1) * (DV_ * 32))
                     : (Kb  + (size_t)(t + 1) * (32 * D_))) + half * 2048;
            unsigned dst = (sel ? lVt_off : lK_off)
                         + (unsigned)bb * 8192u + (unsigned)half * 4096u;
            tdm_load_4kb(dst, gsrc);
        }
#else
        uint4 regs[8];
        if (have_next) {
            const uint4* kp = (const uint4*)(Kb  + (size_t)(t + 1) * 32 * D_);
            const uint4* vp = (const uint4*)(Vtb + (size_t)(t + 1) * (DV_ * 32));
            #pragma unroll
            for (int i = 0; i < 4; ++i) { regs[i] = kp[i * 128 + tid]; regs[4 + i] = vp[i * 128 + tid]; }
        }
#endif

        tile_body<false>(t, lK[t & 1], lVt[t & 1], lP, qa, ones,
                         o, mrow, lrow, wave, lane, vlen);

#if HAVE_TDM
        __builtin_amdgcn_s_wait_tensorcnt(0);   // next tile resident in LDS
        __syncthreads();
#else
        __syncthreads();
        if (have_next) {
            uint4* kd = (uint4*)&lK[(t + 1) & 1][0];
            uint4* vd = (uint4*)&lVt[(t + 1) & 1][0];
            #pragma unroll
            for (int i = 0; i < 4; ++i) { kd[i * 128 + tid] = regs[i]; vd[i * 128 + tid] = regs[4 + i]; }
        }
        __syncthreads();
#endif
    }

    if (rem) {
        tile_body<true>(nfull, lK[nfull & 1], lVt[nfull & 1], lP, qa, ones,
                        o, mrow, lrow, wave, lane, vlen);
    }

    // ---- normalize + store -----------------------------------------------
    float inv[8];
    #pragma unroll
    for (int r = 0; r < 8; ++r) inv[r] = 1.0f / lrow[r];

    float* Ob = Out + ((size_t)b * TQ_ + (size_t)qtile * 64 + (size_t)wave * 16) * DV_;
    #pragma unroll
    for (int c = 0; c < 8; ++c) {
        #pragma unroll
        for (int r = 0; r < 8; ++r) {
            int row = r + ((lane >> 4) << 3);
            Ob[row * DV_ + c * 16 + (lane & 15)] = o[c][r] * inv[r];
        }
    }
}

// ---------------------------------------------------------------------------
// Fallback (self-contained, converts K/V in-loop) for small ws_size.
// ---------------------------------------------------------------------------
__global__ __launch_bounds__(128, 1)
void fattn_fallback(const float* __restrict__ Q, const float* __restrict__ K,
                    const float* __restrict__ V, const int* __restrict__ vlens,
                    float* __restrict__ Out)
{
    __shared__ __align__(16) unsigned short lQ[64 * D_];
    __shared__ __align__(16) unsigned short lKf[32 * D_];
    __shared__ __align__(16) unsigned short lVtf[DV_ * 32];
    __shared__ __align__(16) unsigned short lP[4 * 16 * 32];

    const int b     = blockIdx.y;
    const int qtile = blockIdx.x;
    const int tid   = threadIdx.x;
    const int wave  = tid >> 5;
    const int lane  = tid & 31;

    const float SCALE = 0.08838834764831845f;

    const float* Qb = Q + ((size_t)b * TQ_ + (size_t)qtile * 64) * D_;
    const float* Kb = K + (size_t)b * TK_ * D_;
    const float* Vb = V + (size_t)b * TK_ * DV_;

    for (int i = 0; i < 16; ++i) {
        int idx4 = i * 128 + tid;
        float4 f = ((const float4*)Qb)[idx4];
        int h = idx4 * 4;
        *(unsigned*)&lQ[h]     = packbf(f.x * SCALE, f.y * SCALE);
        *(unsigned*)&lQ[h + 2] = packbf(f.z * SCALE, f.w * SCALE);
    }
    __syncthreads();

    Frag qa[4];
    {
        int m  = lane & 15;
        int kb = (lane < 16) ? 0 : 8;
        for (int c = 0; c < 4; ++c) {
            int base = (wave * 16 + m) * D_ + c * 32;
            #pragma unroll
            for (int j = 0; j < 8; ++j) {
                int k = (j < 4) ? (kb + 2 * j) : (16 + kb + 2 * (j - 4));
                qa[c].u[j] = *(const unsigned*)&lQ[base + k];
            }
        }
    }

    Frag ones;
    #pragma unroll
    for (int j = 0; j < 8; ++j) ones.u[j] = 0x3F803F80u;

    f32x8 o[8];
    float mrow[8], lrow[8];
    #pragma unroll
    for (int c = 0; c < 8; ++c)
        #pragma unroll
        for (int r = 0; r < 8; ++r) o[c][r] = 0.0f;
    #pragma unroll
    for (int r = 0; r < 8; ++r) { mrow[r] = -3.0e38f; lrow[r] = 0.0f; }

    int vlen = vlens[b];
    if (vlen < 1) vlen = 1;
    if (vlen > TK_) vlen = TK_;
    const int nfull  = vlen >> 5;
    const int rem    = vlen & 31;
    const int ntiles = nfull + (rem ? 1 : 0);

    for (int t = 0; t < ntiles; ++t) {
        __syncthreads();
        const float* Kt = Kb + (size_t)t * 32 * D_;
        const float* Vt = Vb + (size_t)t * 32 * DV_;
        for (int i = 0; i < 8; ++i) {
            int idx4 = i * 128 + tid;
            float4 fk = ((const float4*)Kt)[idx4];
            float4 fv = ((const float4*)Vt)[idx4];
            int h   = idx4 * 4;
            int key = h >> 7;
            int dim = h & 127;
            *(unsigned*)&lKf[h]     = packbf(fk.x, fk.y);
            *(unsigned*)&lKf[h + 2] = packbf(fk.z, fk.w);
            lVtf[(dim + 0) * 32 + key] = f2bf(fv.x);
            lVtf[(dim + 1) * 32 + key] = f2bf(fv.y);
            lVtf[(dim + 2) * 32 + key] = f2bf(fv.z);
            lVtf[(dim + 3) * 32 + key] = f2bf(fv.w);
        }
        __syncthreads();

        if (t < nfull)
            tile_body<false>(t, lKf, lVtf, lP, qa, ones, o, mrow, lrow, wave, lane, vlen);
        else
            tile_body<true>(t, lKf, lVtf, lP, qa, ones, o, mrow, lrow, wave, lane, vlen);
    }

    float inv[8];
    #pragma unroll
    for (int r = 0; r < 8; ++r) inv[r] = 1.0f / lrow[r];

    float* Ob = Out + ((size_t)b * TQ_ + (size_t)qtile * 64 + (size_t)wave * 16) * DV_;
    #pragma unroll
    for (int c = 0; c < 8; ++c) {
        #pragma unroll
        for (int r = 0; r < 8; ++r) {
            int row = r + ((lane >> 4) << 3);
            Ob[row * DV_ + c * 16 + (lane & 15)] = o[c][r] * inv[r];
        }
    }
}

extern "C" void kernel_launch(void* const* d_in, const int* in_sizes, int n_in,
                              void* d_out, int out_size, void* d_ws, size_t ws_size,
                              hipStream_t stream) {
    (void)in_sizes; (void)n_in; (void)out_size;
    const float* q  = (const float*)d_in[0];
    const float* k  = (const float*)d_in[1];
    const float* v  = (const float*)d_in[2];
    const int*   vl = (const int*)d_in[3];
    float* out = (float*)d_out;

    const size_t kv_half_bytes = (size_t)B_ * TK_ * D_ * sizeof(unsigned short); // 8 MiB
    dim3 grid(TQ_ / 64, B_);
    dim3 block(128);

    if (ws_size >= 2 * kv_half_bytes) {
        unsigned short* kbf  = (unsigned short*)d_ws;
        unsigned short* vtbf = kbf + (size_t)B_ * TK_ * D_;

        int n4 = B_ * TK_ * D_ / 4;
        hipLaunchKernelGGL(conv_k_kernel, dim3((n4 + 255) / 256), dim3(256), 0, stream,
                           k, kbf, n4);
        hipLaunchKernelGGL(conv_vt_kernel, dim3(TK_ / 32, B_), dim3(256), 0, stream,
                           v, vtbf);
        hipLaunchKernelGGL(fattn_fast, grid, block, 0, stream, q, kbf, vtbf, vl, out);
    } else {
        hipLaunchKernelGGL(fattn_fallback, grid, block, 0, stream, q, k, v, vl, out);
    }
}
